// RnnModel_10797547782344
// MI455X (gfx1250) — matrix operand order repaired and checked by reading, not compiled, not run on hardware
//
#include <hip/hip_runtime.h>
#include <math.h>

// ---------------------------------------------------------------------------
// CDNA5 (gfx1250, wave32). All GEMMs via v_wmma_f32_16x16x32_bf16, f32 accum.
// ---------------------------------------------------------------------------

typedef __attribute__((ext_vector_type(16))) __bf16 bf16x16;
typedef __attribute__((ext_vector_type(8)))  float  f32x8;

#define EPI_STORE      0   // C = acc
#define EPI_BIAS       1   // C = acc + bias[n]
#define EPI_BIAS_PRELU 2   // C = prelu(acc + bias[n], a)
#define EPI_LOGITS     3   // logits[m] += sum_n prelu(acc+bias[n],a) * w2[n]

#define NB   512
#define NA   256
#define NC   128
#define ND   300
#define NH   300
#define NV   10002
#define BD   (NB*ND)
#define BH   (NB*NH)

__device__ __forceinline__ unsigned pk_bf16(float a, float b) {
  union { __bf16 h[2]; unsigned u; } x;
  x.h[0] = (__bf16)a; x.h[1] = (__bf16)b;
  return x.u;                                     // -> v_cvt_pk_bf16_f32
}

// ---------------------------------------------------------------------------
// Tiled GEMM: C[M,N] = A[M,K] @ B[K,N], M must be a multiple of 64 (all uses are).
//   bIsWeight==1: Bsrc row-major (N,K) (torch weight, B = W^T); ==0: (K,N).
// Block: 256 thr = 8 waves -> 64(M) x 64(N) tile, K by 32; 2 WMMA / wave / chunk.
// Register-prefetch software pipeline; uniform fast/slow staging path per chunk.
// ---------------------------------------------------------------------------
__global__ __launch_bounds__(256) void wmma_gemm_bf16(
    const float* __restrict__ A, const float* __restrict__ Bsrc, float* __restrict__ Cout,
    int M, int N, int K, int lda, int ldb, int ldc,
    long sA, long sB, long sC,
    int bIsWeight, int epi,
    const float* __restrict__ bias, const float* __restrict__ preluA,
    const float* __restrict__ w2, float* __restrict__ logitsOut)
{
  __shared__ unsigned AsU[64 * 17];   // bf16[64][34]: row stride 17 dwords
  __shared__ unsigned BsU[64 * 17];   // B^T: Bs[n][k]
  __bf16* Bsh = (__bf16*)BsU;

  const int tid   = threadIdx.x;
  const int mBase = blockIdx.y * 64;
  const int nBase = blockIdx.x * 64;
  const long aOff = (long)blockIdx.z * sA;
  const long bOff = (long)blockIdx.z * sB;
  const long cOff = (long)blockIdx.z * sC;

  const int wave = tid >> 5, wm = wave >> 1, wn = wave & 1;
  const int lane = tid & 31, lhalf = lane >> 4, l15 = lane & 15;

  f32x8 acc0 = {0,0,0,0,0,0,0,0};
  f32x8 acc1 = {0,0,0,0,0,0,0,0};

  float4 ra[2], rb[2];

  // ---- stage chunk k0 into registers (uniform branch; unconditional clamped loads)
  auto loadAB = [&](int k0) {
    const bool fullK = (k0 + 32) <= K;
    const bool fullN = (nBase + 64) <= N;
    #pragma unroll
    for (int i = 0; i < 2; ++i) {
      const int e = tid + i * 256;                // 0..511
      // A: 64 rows x 32 k  (row = e>>3, k group = e&7)
      {
        const int r = e >> 3, q = e & 7;
        const int gk = k0 + q * 4;
        const float* Ap = A + aOff + (long)(mBase + r) * lda;
        if (fullK) {
          ra[i] = *reinterpret_cast<const float4*>(Ap + gk);
        } else {
          float4 v;
          v.x = Ap[min(gk + 0, K - 1)]; if (gk + 0 >= K) v.x = 0.f;
          v.y = Ap[min(gk + 1, K - 1)]; if (gk + 1 >= K) v.y = 0.f;
          v.z = Ap[min(gk + 2, K - 1)]; if (gk + 2 >= K) v.z = 0.f;
          v.w = Ap[min(gk + 3, K - 1)]; if (gk + 3 >= K) v.w = 0.f;
          ra[i] = v;
        }
      }
      // B: 64 n x 32 k
      if (bIsWeight) {                            // W[n][k]: n = e>>3, k group = e&7
        const int r = e >> 3, q = e & 7;
        const int gn = nBase + r, gk = k0 + q * 4;
        if (fullK && fullN) {
          rb[i] = *reinterpret_cast<const float4*>(Bsrc + bOff + (long)gn * ldb + gk);
        } else {
          const float* Bp = Bsrc + bOff + (long)min(gn, N - 1) * ldb;
          const bool nok = gn < N;
          float4 v;
          v.x = Bp[min(gk + 0, K - 1)]; if (!nok || gk + 0 >= K) v.x = 0.f;
          v.y = Bp[min(gk + 1, K - 1)]; if (!nok || gk + 1 >= K) v.y = 0.f;
          v.z = Bp[min(gk + 2, K - 1)]; if (!nok || gk + 2 >= K) v.z = 0.f;
          v.w = Bp[min(gk + 3, K - 1)]; if (!nok || gk + 3 >= K) v.w = 0.f;
          rb[i] = v;
        }
      } else {                                    // B[k][n]: k = e>>4, n group = e&15
        const int kr = e >> 4, nq = e & 15;
        const int gk = k0 + kr, gn = nBase + nq * 4;
        if (fullK && fullN) {
          rb[i] = *reinterpret_cast<const float4*>(Bsrc + bOff + (long)gk * ldb + gn);
        } else {
          const float* Bp = Bsrc + bOff + (long)min(gk, K - 1) * ldb;
          const bool kok = gk < K;
          float4 v;
          v.x = Bp[min(gn + 0, N - 1)]; if (!kok || gn + 0 >= N) v.x = 0.f;
          v.y = Bp[min(gn + 1, N - 1)]; if (!kok || gn + 1 >= N) v.y = 0.f;
          v.z = Bp[min(gn + 2, N - 1)]; if (!kok || gn + 2 >= N) v.z = 0.f;
          v.w = Bp[min(gn + 3, N - 1)]; if (!kok || gn + 3 >= N) v.w = 0.f;
          rb[i] = v;
        }
      }
    }
  };

  // ---- write staged registers to LDS as bf16
  auto storeAB = [&]() {
    #pragma unroll
    for (int i = 0; i < 2; ++i) {
      const int e = tid + i * 256;
      { const int r = e >> 3, q = e & 7;
        AsU[r * 17 + q * 2 + 0] = pk_bf16(ra[i].x, ra[i].y);
        AsU[r * 17 + q * 2 + 1] = pk_bf16(ra[i].z, ra[i].w); }
      if (bIsWeight) {
        const int r = e >> 3, q = e & 7;
        BsU[r * 17 + q * 2 + 0] = pk_bf16(rb[i].x, rb[i].y);
        BsU[r * 17 + q * 2 + 1] = pk_bf16(rb[i].z, rb[i].w);
      } else {                                    // transpose to Bs[n][k]
        const int kr = e >> 4, nq = e & 15;
        Bsh[(nq * 4 + 0) * 34 + kr] = (__bf16)rb[i].x;
        Bsh[(nq * 4 + 1) * 34 + kr] = (__bf16)rb[i].y;
        Bsh[(nq * 4 + 2) * 34 + kr] = (__bf16)rb[i].z;
        Bsh[(nq * 4 + 3) * 34 + kr] = (__bf16)rb[i].w;
      }
    }
  };

  loadAB(0);
  for (int k0 = 0; k0 < K; k0 += 32) {
    storeAB();
    __syncthreads();
    if (k0 + 32 < K) loadAB(k0 + 32);            // prefetch: overlaps ds reads + WMMA

    // fragment gather: contiguous k = 16*lhalf + s in A and B
    union { unsigned u[8]; bf16x16 v; } af, bf0, bf1;
    const unsigned* aRow  = AsU + (wm * 16 + l15) * 17;
    #pragma unroll
    for (int j = 0; j < 4; ++j) { af.u[j] = aRow[8 * lhalf + j]; af.u[4 + j] = aRow[4 + 8 * lhalf + j]; }
    const unsigned* bRow0 = BsU + (wn * 32 + l15) * 17;
    const unsigned* bRow1 = BsU + (wn * 32 + 16 + l15) * 17;
    #pragma unroll
    for (int j = 0; j < 8; ++j) { bf0.u[j] = bRow0[8 * lhalf + j]; bf1.u[j] = bRow1[8 * lhalf + j]; }

    acc0 = __builtin_amdgcn_wmma_f32_16x16x32_bf16(false, af.v, false, bf0.v, (short)0, acc0, false, false);
    acc1 = __builtin_amdgcn_wmma_f32_16x16x32_bf16(false, af.v, false, bf1.v, (short)0, acc1, false, false);
    __syncthreads();
  }

  // ---- epilogue: n = l15(+16), m = row0 + r   (M % 64 == 0 -> no row guard)
  const int col0 = nBase + wn * 32 + l15;
  const int col1 = col0 + 16;
  const int row0 = mBase + wm * 16 + lhalf * 8;

  if (epi == EPI_LOGITS) {
    const float a_pre = preluA[0];
    const bool c0 = col0 < N, c1 = col1 < N;
    const float bv0 = c0 ? bias[col0] : 0.f, wv0 = c0 ? w2[col0] : 0.f;
    const float bv1 = c1 ? bias[col1] : 0.f, wv1 = c1 ? w2[col1] : 0.f;
    #pragma unroll
    for (int r = 0; r < 8; ++r) {
      float v0 = acc0[r] + bv0; v0 = (v0 >= 0.f) ? v0 : a_pre * v0;
      float v1 = acc1[r] + bv1; v1 = (v1 >= 0.f) ? v1 : a_pre * v1;
      float t = (c0 ? v0 * wv0 : 0.f) + (c1 ? v1 * wv1 : 0.f);
      t += __shfl_xor(t, 1, 32);
      t += __shfl_xor(t, 2, 32);
      t += __shfl_xor(t, 4, 32);
      t += __shfl_xor(t, 8, 32);
      if (l15 == 0) atomicAdd(&logitsOut[row0 + r], t);
    }
  } else {
    const bool c0 = col0 < N, c1 = col1 < N;
    const float bv0 = (epi >= EPI_BIAS && c0) ? bias[col0] : 0.f;
    const float bv1 = (epi >= EPI_BIAS && c1) ? bias[col1] : 0.f;
    const float a_pre = (epi == EPI_BIAS_PRELU) ? preluA[0] : 0.f;
    #pragma unroll
    for (int r = 0; r < 8; ++r) {
      const long rowOff = cOff + (long)(row0 + r) * ldc;
      if (c0) {
        float v = acc0[r] + bv0;
        if (epi == EPI_BIAS_PRELU) v = (v >= 0.f) ? v : a_pre * v;
        Cout[rowOff + col0] = v;
      }
      if (c1) {
        float v = acc1[r] + bv1;
        if (epi == EPI_BIAS_PRELU) v = (v >= 0.f) ? v : a_pre * v;
        Cout[rowOff + col1] = v;
      }
    }
  }
}

// ---------------------------------------------------------------------------
// Pointwise / gather / reduction kernels (f32)
// ---------------------------------------------------------------------------
__global__ void init_kernel(float* __restrict__ logits, float* __restrict__ out)
{
  int i = blockIdx.x * 256 + threadIdx.x;
  if (i < NB * NC) logits[i] = 0.f;
  if (i < 2) out[i] = 0.f;
}

__global__ void gather_x_kernel(const float* __restrict__ rc, const float* __restrict__ emb,
                                const int* __restrict__ seq, const int* __restrict__ cidx,
                                float* __restrict__ xbuf)
{
  int idx = blockIdx.x * 256 + threadIdx.x;
  if (idx >= BD) return;
  int b = idx / ND, d = idx % ND;
  xbuf[idx]          = rc[((long)b * NC + cidx[b]) * ND + d];
  xbuf[BD + idx]     = emb[(long)seq[b * 4 + 1] * ND + d];
  xbuf[2 * BD + idx] = emb[(long)seq[b * 4 + 2] * ND + d];
}

__global__ void gru_gates_kernel(const float* __restrict__ gi, const float* __restrict__ gh,
                                 const float* __restrict__ hprev, float* __restrict__ hout)
{
  int idx = blockIdx.x * 256 + threadIdx.x;
  if (idx >= BH) return;
  int b = idx / NH, j = idx % NH;
  const float* gib = gi + (long)b * (3 * NH);
  const float* ghb = gh + (long)b * (3 * NH);
  float r = 1.f / (1.f + expf(-(gib[j] + ghb[j])));
  float z = 1.f / (1.f + expf(-(gib[NH + j] + ghb[NH + j])));
  float n = tanhf(gib[2 * NH + j] + r * ghb[2 * NH + j]);
  hout[idx] = (1.f - z) * n + z * hprev[idx];
}

__global__ void gather_out2_kernel(const float* __restrict__ y1, float* __restrict__ out2)
{
  int idx = blockIdx.x * 256 + threadIdx.x;
  if (idx >= 1024 * NH) return;
  int row = idx / NH, j = idx % NH;
  int b = row >> 1, t = row & 1;
  out2[idx] = y1[((long)t * NB + b) * NH + j];
}

__global__ void loss_prod_kernel(const float* __restrict__ logits, const float* __restrict__ labels,
                                 const float* __restrict__ bp2, float* __restrict__ out)
{
  __shared__ float sv[NC]; __shared__ int si[NC];
  __shared__ float sm[NC]; __shared__ float ss[NC];
  int b = blockIdx.x, c = threadIdx.x;
  float lg = logits[b * NC + c] + bp2[0];
  sv[c] = lg; si[c] = c; __syncthreads();
  for (int s = NC / 2; s > 0; s >>= 1) {
    if (c < s) {
      float v2 = sv[c + s]; int i2 = si[c + s];
      if (v2 > sv[c] || (v2 == sv[c] && i2 < si[c])) { sv[c] = v2; si[c] = i2; }
    }
    __syncthreads();
  }
  float mx = sv[0]; int amax = si[0];
  __syncthreads();
  float lab = labels[b * NC + c];
  float lex = expf(lg - mx);
  sm[c] = (lab > 0.f) ? lex : __builtin_inff();
  ss[c] = lex * (1.f - lab);
  __syncthreads();
  for (int s = NC / 2; s > 0; s >>= 1) {
    if (c < s) { sm[c] = fminf(sm[c], sm[c + s]); ss[c] += ss[c + s]; }
    __syncthreads();
  }
  if (c == 0) {
    float prob = sm[0] / (ss[0] + sm[0] + 1e-6f);
    atomicAdd(&out[0], -logf(prob + 1e-6f) * (1.0f / (float)NB));
    out[2 + b] = labels[b * NC + amax];
  }
}

__global__ void loss_react_kernel(const float* __restrict__ rl, const int* __restrict__ seq,
                                  float* __restrict__ out)
{
  __shared__ float sv[256]; __shared__ int si[256]; __shared__ float ssum[256];
  int row = blockIdx.x;                 // 0..1023 = b*2 + t
  int b = row >> 1, t = row & 1;
  int target = seq[b * 4 + 1 + t];
  const float* x = rl + (long)row * NV;
  int tid = threadIdx.x;

  float bm = -__builtin_inff(); int bi = NV;
  for (int i = tid; i < NV; i += 256) {
    float v = x[i];
    if (v > bm || (v == bm && i < bi)) { bm = v; bi = i; }
  }
  sv[tid] = bm; si[tid] = bi; __syncthreads();
  for (int s = 128; s > 0; s >>= 1) {
    if (tid < s) {
      float v2 = sv[tid + s]; int i2 = si[tid + s];
      if (v2 > sv[tid] || (v2 == sv[tid] && i2 < si[tid])) { sv[tid] = v2; si[tid] = i2; }
    }
    __syncthreads();
  }
  float mx = sv[0]; int amax = si[0];
  __syncthreads();
  float se = 0.f;
  for (int i = tid; i < NV; i += 256) se += expf(x[i] - mx);
  ssum[tid] = se; __syncthreads();
  for (int s = 128; s > 0; s >>= 1) { if (tid < s) ssum[tid] += ssum[tid + s]; __syncthreads(); }
  if (tid == 0) {
    float lse = mx + logf(ssum[0]);
    float lp  = x[target] - lse;
    atomicAdd(&out[1], -lp * (1.0f / 1024.0f));
    out[2 + NB + row] = (amax == target) ? 1.0f : 0.0f;
  }
}

// ---------------------------------------------------------------------------
// Host orchestration
// ---------------------------------------------------------------------------
extern "C" void kernel_launch(void* const* d_in, const int* in_sizes, int n_in,
                              void* d_out, int out_size, void* d_ws, size_t ws_size,
                              hipStream_t stream)
{
  (void)in_sizes; (void)n_in; (void)out_size; (void)ws_size;
  const float* atoms = (const float*)d_in[0];
  const float* mols  = (const float*)d_in[1];
  const float* aidx  = (const float*)d_in[2];
  const float* labels= (const float*)d_in[3];
  const int*   seq   = (const int*)d_in[4];
  const int*   cidx  = (const int*)d_in[5];
  const float* emb   = (const float*)d_in[6];
  const float* Wm    = (const float*)d_in[7];
  const float* bm    = (const float*)d_in[8];
  const float* am    = (const float*)d_in[9];
  const float* Wp1   = (const float*)d_in[10];
  const float* bp1   = (const float*)d_in[11];
  const float* ap1   = (const float*)d_in[12];
  const float* Wp2   = (const float*)d_in[13];
  const float* bp2   = (const float*)d_in[14];
  const float* Wr1   = (const float*)d_in[15];
  const float* br1   = (const float*)d_in[16];
  const float* ar    = (const float*)d_in[17];
  const float* Wr2   = (const float*)d_in[18];
  const float* br2   = (const float*)d_in[19];
  const float* Wih0  = (const float*)d_in[20];
  const float* Whh0  = (const float*)d_in[21];
  const float* bih0  = (const float*)d_in[22];
  const float* bhh0  = (const float*)d_in[23];
  const float* Wih1  = (const float*)d_in[24];
  const float* Whh1  = (const float*)d_in[25];
  const float* bih1  = (const float*)d_in[26];
  const float* bhh1  = (const float*)d_in[27];
  float* out = (float*)d_out;

  char* ws = (char*)d_ws;
  size_t off = 0;
  auto alloc = [&](size_t elems) -> float* {
    float* p = (float*)(ws + off);
    off = (off + elems * sizeof(float) + 255) & ~(size_t)255;
    return p;
  };
  float* rc     = alloc((size_t)NB * NC * ND);
  float* logits = alloc((size_t)NB * NC);
  float* xbuf   = alloc((size_t)3 * BD);
  float* h0buf  = alloc((size_t)NB * 2 * NH);
  float* gib    = alloc((size_t)NB * 3 * NH);
  float* ghb    = alloc((size_t)NB * 3 * NH);
  float* y0     = alloc((size_t)3 * BH);
  float* y1     = alloc((size_t)3 * BH);
  float* out2   = alloc((size_t)1024 * NH);
  float* hidr   = alloc((size_t)1024 * 2 * NH);
  float* rl     = alloc((size_t)1024 * NV);

  auto gemm = [&](const float* A, const float* Bm_, float* C,
                  int M, int N, int K, int lda, int ldb, int ldc,
                  long sA, long sB, long sC, int nbatch,
                  int bW, int epi, const float* bias, const float* pa,
                  const float* w2, float* lg) {
    dim3 grid((N + 63) / 64, M / 64, nbatch);
    wmma_gemm_bf16<<<grid, dim3(256), 0, stream>>>(A, Bm_, C, M, N, K, lda, ldb, ldc,
                                                   sA, sB, sC, bW, epi, bias, pa, w2, lg);
  };

  init_kernel<<<256, 256, 0, stream>>>(logits, out);

  // 1) rc[b] = atom_indexes[b] (128x256) @ atoms[b] (256x300)
  gemm(aidx, atoms, rc, NC, ND, NA, NA, ND, ND,
       (long)NC * NA, (long)NA * ND, (long)NC * ND, NB,
       0, EPI_STORE, nullptr, nullptr, nullptr, nullptr);

  // 2) logits[b,c] = prelu(rc @ Wp1.T + bp1, ap1) . Wp2  (fused, atomic row-sum)
  gemm(rc, Wp1, nullptr, NB * NC, ND, ND, ND, ND, 0,
       0, 0, 0, 1, 1, EPI_LOGITS, bp1, ap1, Wp2, logits);

  // 3) GRU inputs x (3,B,D)
  gather_x_kernel<<<(BD + 255) / 256, 256, 0, stream>>>(rc, emb, seq, cidx, xbuf);

  // 4) h0 = prelu(mols @ Wm.T + bm, am) -> contiguous (B,600); layer-l h0 = +l*B*H
  gemm(mols, Wm, h0buf, NB, 2 * NH, ND, ND, ND, 2 * NH,
       0, 0, 0, 1, 1, EPI_BIAS_PRELU, bm, am, nullptr, nullptr);

  // 5) GRU layers (h_new == ys[t] -> hidden state aliases y buffers)
  const float* h = h0buf;
  for (int t = 0; t < 3; ++t) {
    gemm(xbuf + (size_t)t * BD, Wih0, gib, NB, 3 * NH, ND, ND, ND, 3 * NH,
         0, 0, 0, 1, 1, EPI_BIAS, bih0, nullptr, nullptr, nullptr);
    gemm(h, Whh0, ghb, NB, 3 * NH, NH, NH, NH, 3 * NH,
         0, 0, 0, 1, 1, EPI_BIAS, bhh0, nullptr, nullptr, nullptr);
    gru_gates_kernel<<<(BH + 255) / 256, 256, 0, stream>>>(gib, ghb, h, y0 + (size_t)t * BH);
    h = y0 + (size_t)t * BH;
  }
  h = h0buf + (size_t)BH;
  for (int t = 0; t < 3; ++t) {
    gemm(y0 + (size_t)t * BH, Wih1, gib, NB, 3 * NH, NH, NH, NH, 3 * NH,
         0, 0, 0, 1, 1, EPI_BIAS, bih1, nullptr, nullptr, nullptr);
    gemm(h, Whh1, ghb, NB, 3 * NH, NH, NH, NH, 3 * NH,
         0, 0, 0, 1, 1, EPI_BIAS, bhh1, nullptr, nullptr, nullptr);
    gru_gates_kernel<<<(BH + 255) / 256, 256, 0, stream>>>(gib, ghb, h, y1 + (size_t)t * BH);
    h = y1 + (size_t)t * BH;
  }

  // 6) react head (t=0,1 only, i.e. rl[:, :-1])
  gather_out2_kernel<<<(1024 * NH + 255) / 256, 256, 0, stream>>>(y1, out2);
  gemm(out2, Wr1, hidr, 1024, 2 * NH, NH, NH, NH, 2 * NH,
       0, 0, 0, 1, 1, EPI_BIAS_PRELU, br1, ar, nullptr, nullptr);
  gemm(hidr, Wr2, rl, 1024, NV, 2 * NH, 2 * NH, 2 * NH, NV,
       0, 0, 0, 1, 1, EPI_BIAS, br2, nullptr, nullptr, nullptr);

  // 7) losses + argmax outputs
  loss_prod_kernel<<<NB, NC, 0, stream>>>(logits, labels, bp2, out);
  loss_react_kernel<<<1024, 256, 0, stream>>>(rl, seq, out);
}
